// SliceSEFusionFixedWindow_55233279427198
// MI455X (gfx1250) — matrix operand order repaired and checked by reading, not compiled, not run on hardware
//
#include <hip/hip_runtime.h>
#include <hip/hip_bf16.h>

// Problem constants (from reference setup)
#define BB   2
#define SS   5
#define CC   256
#define HH   192
#define WW   192
#define KWIN 7
#define NH   28
#define NW   28
#define HID  16

typedef __attribute__((ext_vector_type(16))) _Float16 v16h;
typedef __attribute__((ext_vector_type(8)))  float    v8f;

// ---------------------------------------------------------------------------
// Kernel 1: 7x7 window means over padded x.
// One block per (b, s, c, nh) strip: stream 7 x 192 contiguous floats,
// bin into 28 window sums via LDS float atomics (ds_add_f32).
// g layout: [b][nh][nw][s][c]  (so kernel 2 reads contiguous rows)
// ---------------------------------------------------------------------------
__global__ __launch_bounds__(256) void sse_window_means(
    const float* __restrict__ x, float* __restrict__ g)
{
    int bid = blockIdx.x;                 // ((b*SS + s)*CC + c)*NH + nh
    int nh  = bid % NH;  int t = bid / NH;
    int c   = t % CC;    t /= CC;
    int s   = t % SS;    int b = t / SS;

    __shared__ float sums[NW];
    int tid = threadIdx.x;
    if (tid < NW) sums[tid] = 0.0f;
    __syncthreads();

    const float* xr = x + ((((size_t)b * SS + s) * CC + c) * HH) * (size_t)WW;
    // 7 rows x 192 cols, fully coalesced (consecutive tid -> consecutive col)
    for (int idx = tid; idx < KWIN * WW; idx += 256) {
        int r   = idx / WW;
        int col = idx % WW;
        int hi  = nh * KWIN + r;          // padded rows >= HH contribute 0
        if (hi < HH) {
            atomicAdd(&sums[col / KWIN], xr[(size_t)hi * WW + col]);
        }
    }
    __syncthreads();
    if (tid < NW) {
        size_t gi = ((((size_t)b * NH + nh) * NW + tid) * SS + s) * CC + c;
        g[gi] = sums[tid] * (1.0f / (KWIN * KWIN));  // mean incl. zero padding
    }
}

// ---------------------------------------------------------------------------
// Kernel 2: tiny MLP (C=256 -> 16 -> 1) + static logits + softmax over S.
// One wave (32 threads) per window position. The 256->16 projection runs on
// v_wmma_f32_16x16x32_f16: M=16 (S padded to 16), N=16 (hidden), K=256 (8
// chunks of 32). Fragment layouts per CDNA5 ISA 7.12.2.
// ---------------------------------------------------------------------------
__global__ __launch_bounds__(32) void sse_mlp_softmax(
    const float* __restrict__ g,
    const float* __restrict__ w1,   // [16][256]
    const float* __restrict__ b1,   // [16]
    const float* __restrict__ w2,   // [1][16]
    const float* __restrict__ b2,   // [1]
    const float* __restrict__ stat, // [5]
    float* __restrict__ wt)         // [b][nh][nw][s]
{
    const int pos = blockIdx.x;           // b*NH*NW + nh*NW + nw
    const int l   = threadIdx.x;          // 0..31
    const int m   = l & 15;               // A-row / B-col / D-col index
    const int hs  = l >> 4;               // half-wave select

    const float* gp = g + (size_t)pos * SS * CC;   // [s][c], row stride CC

    __shared__ float hsh[16][17];
    __shared__ float lg[8];

    v8f acc = {};
    #pragma unroll
    for (int ck = 0; ck < CC / 32; ++ck) {
        const int k0 = ck * 32;
        v16h a, bf;
        // A fragment: row m, lane holds K = k0+hs*8+i (i<8) and k0+16+hs*8+i
        if (m < SS) {
            const float* gr = gp + (size_t)m * CC + k0;
            #pragma unroll
            for (int i = 0; i < 8; ++i) a[i]     = (_Float16)gr[hs * 8 + i];
            #pragma unroll
            for (int i = 0; i < 8; ++i) a[8 + i] = (_Float16)gr[16 + hs * 8 + i];
        } else {
            #pragma unroll
            for (int i = 0; i < 16; ++i) a[i] = (_Float16)0.0f;
        }
        // B fragment: column n = m (hidden idx), lane holds K = k0+hs*8+i (i<8) and k0+16+hs*8+i.
        // B[k][n] = w1[n][k]  -> contiguous row of w1.
        const float* wr = w1 + (size_t)m * CC + k0;
        #pragma unroll
        for (int i = 0; i < 16; ++i) bf[i] = (_Float16)wr[(i & 8) * 2 + hs * 8 + (i & 7)];

        acc = __builtin_amdgcn_wmma_f32_16x16x32_f16(
            /*neg_a=*/false, a, /*neg_b=*/false, bf,
            /*c_mod=*/(short)0, acc, /*reuse_a=*/false, /*reuse_b=*/false);
    }

    // D layout: lane l holds rows (v + 8*hs), column n = l&15.  Bias + ReLU.
    #pragma unroll
    for (int v = 0; v < 8; ++v) {
        int   md = v + 8 * hs;
        float hv = acc[v] + b1[m];
        hsh[md][m] = hv > 0.0f ? hv : 0.0f;
    }
    __syncthreads();

    if (l < SS) {
        float lgt = b2[0] + stat[l];
        #pragma unroll
        for (int n = 0; n < HID; ++n) lgt += hsh[l][n] * w2[n];
        lg[l] = lgt;
    }
    __syncthreads();

    if (l < SS) {
        float mx = lg[0];
        #pragma unroll
        for (int s = 1; s < SS; ++s) mx = fmaxf(mx, lg[s]);
        float den = 0.0f;
        #pragma unroll
        for (int s = 0; s < SS; ++s) den += __expf(lg[s] - mx);
        wt[(size_t)pos * SS + l] = __expf(lg[l] - mx) / den;
    }
}

// ---------------------------------------------------------------------------
// Kernel 3: softmax-weighted sum + permuted re-layout.
// out[b,c, kh*28+nh, kw*28+nw] = sum_s wt[b,nh,nw,s] * x_pad[b,s,c, nh*7+kh, nw*7+kw]
// One block per output row (b,c,y): contiguous loads of the 5 needed input
// rows into LDS (zero-padding cols/rows >= 192), stride-7 LDS gather
// (conflict-free: gcd(7,64)=1), coalesced output-row store.
// ---------------------------------------------------------------------------
__global__ __launch_bounds__(192) void sse_weighted_fuse(
    const float* __restrict__ x,
    const float* __restrict__ wt,
    float* __restrict__ out)
{
    int bid = blockIdx.x;                 // (b*CC + c)*HH + y
    int y   = bid % HH;  int t = bid / HH;
    int c   = t % CC;    int b = t / CC;

    int kh = y / NH;                      // y = kh*28 + nh
    int nh = y % NH;
    int hi = nh * KWIN + kh;              // source row in padded image

    __shared__ float rows[SS][200];       // 196 used; 200 for padding slack
    __shared__ float wrow[NW * SS];       // weights for this (b, nh)

    int tid = threadIdx.x;                // 0..191
    if (tid < NW * SS)
        wrow[tid] = wt[(((size_t)b * NH + nh) * NW) * SS + tid];

    #pragma unroll
    for (int s = 0; s < SS; ++s) {
        const float* xr = x + ((((size_t)b * SS + s) * CC + c) * HH) * (size_t)WW;
        for (int col = tid; col < NW * KWIN; col += 192) {   // 196 cols
            rows[s][col] = (hi < HH && col < WW) ? xr[(size_t)hi * WW + col]
                                                 : 0.0f;
        }
    }
    __syncthreads();

    int xo = tid;                         // output col 0..191
    int kw = xo / NH;
    int nw = xo % NH;
    int cp = nw * KWIN + kw;              // source col in padded image
    float acc = 0.0f;
    #pragma unroll
    for (int s = 0; s < SS; ++s)
        acc += wrow[nw * SS + s] * rows[s][cp];

    out[((((size_t)b * CC + c) * HH) + y) * (size_t)WW + xo] = acc;
}

// ---------------------------------------------------------------------------
// Launch
// ---------------------------------------------------------------------------
extern "C" void kernel_launch(void* const* d_in, const int* in_sizes, int n_in,
                              void* d_out, int out_size, void* d_ws, size_t ws_size,
                              hipStream_t stream) {
    (void)in_sizes; (void)n_in; (void)out_size; (void)ws_size;
    const float* x    = (const float*)d_in[0];
    const float* w1   = (const float*)d_in[1];
    const float* b1   = (const float*)d_in[2];
    const float* w2   = (const float*)d_in[3];
    const float* b2   = (const float*)d_in[4];
    const float* stat = (const float*)d_in[5];

    float* g  = (float*)d_ws;                                   // 2*28*28*5*256 f32 ~ 8.0 MB
    float* wt = g + (size_t)BB * NH * NW * SS * CC;             // 7840 f32

    sse_window_means<<<BB * SS * CC * NH, 256, 0, stream>>>(x, g);
    sse_mlp_softmax<<<BB * NH * NW, 32, 0, stream>>>(g, w1, b1, w2, b2, stat, wt);
    sse_weighted_fuse<<<BB * CC * HH, 192, 0, stream>>>(x, wt, (float*)d_out);
}
